// InteractionBlock_85916525789376
// MI455X (gfx1250) — matrix operand order repaired and checked by reading, not compiled, not run on hardware
//
#include <hip/hip_runtime.h>
#include <hip/hip_bf16.h>

#define FDIM 128

typedef __attribute__((ext_vector_type(16))) __bf16 v16bf;
typedef __attribute__((ext_vector_type(8)))  float  v8f;

__device__ __forceinline__ float silu_f(float v) {
    return v / (1.0f + __expf(-v));
}

// Build one 16x32 bf16 A-operand slice for this lane from an fp32 row.
// Lane holds K = {g*8..g*8+7} and {16+g*8..16+g*8+7} relative to kb base.
__device__ __forceinline__ v16bf build_a_bf16(const float* __restrict__ xr, int kb) {
    float4 f0 = *(const float4*)(xr + kb);
    float4 f1 = *(const float4*)(xr + kb + 4);
    float4 f2 = *(const float4*)(xr + kb + 16);
    float4 f3 = *(const float4*)(xr + kb + 20);
    union { v16bf v; __bf16 e[16]; } a;
    a.e[0]=(__bf16)f0.x; a.e[1]=(__bf16)f0.y; a.e[2]=(__bf16)f0.z; a.e[3]=(__bf16)f0.w;
    a.e[4]=(__bf16)f1.x; a.e[5]=(__bf16)f1.y; a.e[6]=(__bf16)f1.z; a.e[7]=(__bf16)f1.w;
    a.e[8]=(__bf16)f2.x; a.e[9]=(__bf16)f2.y; a.e[10]=(__bf16)f2.z; a.e[11]=(__bf16)f2.w;
    a.e[12]=(__bf16)f3.x; a.e[13]=(__bf16)f3.y; a.e[14]=(__bf16)f3.z; a.e[15]=(__bf16)f3.w;
    return a.v;
}

// ---------------------------------------------------------------------------
// Small prep kernels
// ---------------------------------------------------------------------------
__global__ void bfcvt_kernel(const float* __restrict__ in, __bf16* __restrict__ out, int n) {
    int i = blockIdx.x * 256 + threadIdx.x;
    if (i < n) out[i] = (__bf16)in[i];
}

// W3[k=l*128+i][j] = W_bilin[i][j][l]   (W_bilin row-major [128][128][8])
__global__ void w3prep_kernel(const float* __restrict__ wbil, __bf16* __restrict__ w3) {
    int idx = blockIdx.x * 256 + threadIdx.x;   // idx = k*128 + j
    int k = idx >> 7, j = idx & 127;
    int l = k >> 7, i = k & 127;
    w3[idx] = (__bf16)wbil[((size_t)i * 128 + j) * 8 + l];
}

// ---------------------------------------------------------------------------
// Generic dense layer: out[row,n] = (res? res[row,n] : 0) + silu(X@W + b)
// X fp32 [nrows,128], W bf16 [128,128] row-major (k=in, n=out).
// Block = 256 threads = 8 waves; each wave does a 32-row tile (two 16-row
// strips sharing every B tile -> 2 independent WMMAs per LDS B load).
// ---------------------------------------------------------------------------
__global__ __launch_bounds__(256) void dense_silu_kernel(
    const float* __restrict__ X, const __bf16* __restrict__ Wb,
    const float* __restrict__ bias, const float* res, float* out)
{
    __shared__ __bf16 Wl[128 * 136];   // rows padded to 136 elems (272B) -> low bank conflicts
    const int tid = threadIdx.x;
    {
        const unsigned* src = (const unsigned*)Wb;   // 128x64 dwords
        unsigned* dst = (unsigned*)Wl;               // row stride 68 dwords
        #pragma unroll
        for (int it = 0; it < 32; ++it) {
            int d = it * 256 + tid;
            int r = d >> 6, c = d & 63;
            dst[r * 68 + c] = src[d];
        }
    }
    __syncthreads();

    const int wave = tid >> 5, lane = tid & 31;
    const int m = lane & 15, g = lane >> 4;
    const int rowsbase = blockIdx.x * 256 + wave * 32;
    const float* xr0 = X + (size_t)(rowsbase + m) * FDIM;
    const float* xr1 = X + (size_t)(rowsbase + 16 + m) * FDIM;

    v8f acc[2][8];
    const v8f vz = {0.f,0.f,0.f,0.f,0.f,0.f,0.f,0.f};
    #pragma unroll
    for (int s = 0; s < 2; ++s)
        #pragma unroll
        for (int i = 0; i < 8; ++i) acc[s][i] = vz;

    #pragma unroll
    for (int kk = 0; kk < 4; ++kk) {
        const int kb = kk * 32 + g * 8;
        v16bf a0 = build_a_bf16(xr0, kb);
        v16bf a1 = build_a_bf16(xr1, kb);

        const __bf16* wrow = Wl + (size_t)(kk * 32 + g * 8) * 136 + m;
        #pragma unroll
        for (int nt = 0; nt < 8; ++nt) {
            union { v16bf v; __bf16 e[16]; } b;
            #pragma unroll
            for (int s = 0; s < 16; ++s)
                b.e[s] = wrow[((s & 7) + (s >> 3) * 16) * 136 + nt * 16];
            acc[0][nt] = __builtin_amdgcn_wmma_f32_16x16x32_bf16(
                false, a0, false, b.v, (short)0, acc[0][nt], false, false);
            acc[1][nt] = __builtin_amdgcn_wmma_f32_16x16x32_bf16(
                false, a1, false, b.v, (short)0, acc[1][nt], false, false);
        }
    }

    #pragma unroll
    for (int s = 0; s < 2; ++s) {
        #pragma unroll
        for (int nt = 0; nt < 8; ++nt) {
            const int n = nt * 16 + m;
            const float bv = bias[n];
            #pragma unroll
            for (int r = 0; r < 8; ++r) {
                const int row = rowsbase + s * 16 + r + 8 * g;
                const size_t idx = (size_t)row * FDIM + n;
                float v = silu_f(acc[s][nt][r] + bv);
                if (res) v += res[idx];
                out[idx] = v;
            }
        }
    }
}

// ---------------------------------------------------------------------------
// x_kj_g (bf16) = silu-output x_kj (fp32) * (rbf @ W_rbf)
// ---------------------------------------------------------------------------
__global__ __launch_bounds__(256) void mul_rbf_bf16_kernel(
    const float* __restrict__ xkj, const float* __restrict__ rbf,
    const float* __restrict__ Wrbf, __bf16* __restrict__ out)
{
    int idx = blockIdx.x * 256 + threadIdx.x;
    int e = idx >> 7, n = idx & 127;
    float gsum = 0.f;
    #pragma unroll
    for (int r = 0; r < 6; ++r) gsum += rbf[e * 6 + r] * Wrbf[r * 128 + n];
    out[idx] = (__bf16)(xkj[idx] * gsum);
}

// ---------------------------------------------------------------------------
// sbf_p[t,0..7] = sbf[t,:] @ W_sbf   (memory-bound, LDS-tiled for coalescing)
// ---------------------------------------------------------------------------
__global__ __launch_bounds__(256) void sbf_proj_kernel(
    const float* __restrict__ sbf, const float* __restrict__ Wsbf,
    float* __restrict__ sbfp)
{
    __shared__ float sl[256 * 44];
    __shared__ float wl[42 * 8];
    const int tid = threadIdx.x;
    const int base = blockIdx.x * 256;
    for (int i = tid; i < 256 * 42; i += 256) {
        int r = i / 42, c = i - r * 42;
        sl[r * 44 + c] = sbf[(size_t)(base + r) * 42 + c];
    }
    if (tid < 336) wl[tid] = Wsbf[tid];
    __syncthreads();
    float a[8] = {0.f,0.f,0.f,0.f,0.f,0.f,0.f,0.f};
    for (int r = 0; r < 42; ++r) {
        float v = sl[tid * 44 + r];
        #pragma unroll
        for (int l = 0; l < 8; ++l) a[l] += v * wl[r * 8 + l];
    }
    float4* o = (float4*)(sbfp + (size_t)(base + tid) * 8);
    o[0] = make_float4(a[0], a[1], a[2], a[3]);
    o[1] = make_float4(a[4], a[5], a[6], a[7]);
}

// ---------------------------------------------------------------------------
// Bilinear + segment-sum:
//   agg[id_red[w], j] += sum_k (x_kj[id_exp[w], i] * sbf_p[w,l]) * W3[k,j]
// A built in registers (packed-bf16 per-row scale), W3 column-group in LDS
// (staged with async global->LDS loads, ASYNCcnt-tracked).
// Persistent blocks: blockIdx even/odd = column group (64 cols each).
// Each wave handles 32 triplets/batch: every LDS B tile feeds 2 WMMAs.
// ---------------------------------------------------------------------------
__global__ __launch_bounds__(256) void bilinear_scatter_kernel(
    const __bf16* __restrict__ xkjg, const float* __restrict__ sbfp,
    const int* __restrict__ id_exp, const int* __restrict__ id_red,
    const __bf16* __restrict__ w3, float* agg, int ntrip)
{
    extern __shared__ __bf16 Wl[];   // [1024][72] (rows padded: 144B)
    const int tid = threadIdx.x;
    const int cg  = blockIdx.x & 1;
    const int rb0 = blockIdx.x >> 1;
    {
        // async global->LDS staging of this block's 1024x64 bf16 column group
        const unsigned ldsbase = (unsigned)(reinterpret_cast<uintptr_t>(&Wl[0]));
        #pragma unroll
        for (int it = 0; it < 32; ++it) {
            int ch = it * 256 + tid;              // 8192 chunks of 16B
            int r = ch >> 3, c2 = ch & 7;
            unsigned dstoff = ldsbase + (unsigned)(r * 144 + c2 * 16);
            int srcoff = r * 256 + cg * 128 + c2 * 16;
            asm volatile("global_load_async_to_lds_b128 %0, %1, %2 offset:0"
                         :: "v"(dstoff), "v"(srcoff), "s"(w3)
                         : "memory");
        }
        asm volatile("s_wait_asynccnt 0x0" ::: "memory");
    }
    __syncthreads();

    const int wave = tid >> 5, lane = tid & 31;
    const int m = lane & 15, g = lane >> 4;
    const int nbatch = ntrip >> 8;        // 256 triplets per batch (8 waves x 32)
    const int bstride = gridDim.x >> 1;

    const v8f vz = {0.f,0.f,0.f,0.f,0.f,0.f,0.f,0.f};

    for (int batch = rb0; batch < nbatch; batch += bstride) {
        const int tbase = batch * 256 + wave * 32;
        int eid0 = 0, rid0 = 0, eid1 = 0, rid1 = 0;
        if (lane < 16) {
            eid0 = id_exp[tbase + lane];      rid0 = id_red[tbase + lane];
            eid1 = id_exp[tbase + 16 + lane]; rid1 = id_red[tbase + 16 + lane];
        }
        const __bf16* xr0 = xkjg + (size_t)__shfl(eid0, m, 32) * FDIM;
        const __bf16* xr1 = xkjg + (size_t)__shfl(eid1, m, 32) * FDIM;

        __bf16 sv0[8], sv1[8];
        {
            const float* sp0 = sbfp + (size_t)(tbase + m) * 8;
            const float* sp1 = sbfp + (size_t)(tbase + 16 + m) * 8;
            float4 s0 = *(const float4*)sp0;
            float4 s1 = *(const float4*)(sp0 + 4);
            sv0[0]=(__bf16)s0.x; sv0[1]=(__bf16)s0.y; sv0[2]=(__bf16)s0.z; sv0[3]=(__bf16)s0.w;
            sv0[4]=(__bf16)s1.x; sv0[5]=(__bf16)s1.y; sv0[6]=(__bf16)s1.z; sv0[7]=(__bf16)s1.w;
            float4 s2 = *(const float4*)sp1;
            float4 s3 = *(const float4*)(sp1 + 4);
            sv1[0]=(__bf16)s2.x; sv1[1]=(__bf16)s2.y; sv1[2]=(__bf16)s2.z; sv1[3]=(__bf16)s2.w;
            sv1[4]=(__bf16)s3.x; sv1[5]=(__bf16)s3.y; sv1[6]=(__bf16)s3.z; sv1[7]=(__bf16)s3.w;
        }

        v8f acc[2][4];
        #pragma unroll
        for (int s = 0; s < 2; ++s)
            #pragma unroll
            for (int i = 0; i < 4; ++i) acc[s][i] = vz;

        #pragma unroll
        for (int ir = 0; ir < 4; ++ir) {
            union { v16bf v; uint4 q[2]; } xa0, xa1;  // 16x32 x-slices, reused for all 8 l
            xa0.q[0] = *(const uint4*)(xr0 + ir * 32 + g * 8);
            xa0.q[1] = *(const uint4*)(xr0 + ir * 32 + 16 + g * 8);
            xa1.q[0] = *(const uint4*)(xr1 + ir * 32 + g * 8);
            xa1.q[1] = *(const uint4*)(xr1 + ir * 32 + 16 + g * 8);
            #pragma unroll
            for (int l = 0; l < 8; ++l) {
                v16bf av0 = xa0.v * sv0[l];       // packed bf16 per-row scale
                v16bf av1 = xa1.v * sv1[l];
                const int kk = l * 4 + ir;
                const __bf16* wrow = Wl + (size_t)(kk * 32 + g * 8) * 72 + m;
                #pragma unroll
                for (int nt = 0; nt < 4; ++nt) {
                    union { v16bf v; __bf16 e[16]; } b;
                    #pragma unroll
                    for (int s = 0; s < 16; ++s)
                        b.e[s] = wrow[((s & 7) + (s >> 3) * 16) * 72 + nt * 16];
                    acc[0][nt] = __builtin_amdgcn_wmma_f32_16x16x32_bf16(
                        false, av0, false, b.v, (short)0, acc[0][nt], false, false);
                    acc[1][nt] = __builtin_amdgcn_wmma_f32_16x16x32_bf16(
                        false, av1, false, b.v, (short)0, acc[1][nt], false, false);
                }
            }
        }

        #pragma unroll
        for (int s = 0; s < 2; ++s) {
            const int rid = s ? rid1 : rid0;
            #pragma unroll
            for (int nt = 0; nt < 4; ++nt) {
                const int n = cg * 64 + nt * 16 + m;
                #pragma unroll
                for (int r = 0; r < 8; ++r) {
                    const int red = __shfl(rid, r + 8 * g, 32);
                    unsafeAtomicAdd(agg + (size_t)red * FDIM + n, acc[s][nt][r]);
                }
            }
        }
    }
}

// ---------------------------------------------------------------------------
extern "C" void kernel_launch(void* const* d_in, const int* in_sizes, int n_in,
                              void* d_out, int out_size, void* d_ws, size_t ws_size,
                              hipStream_t stream) {
    const float* x      = (const float*)d_in[0];
    const float* rbf    = (const float*)d_in[1];
    const float* sbf    = (const float*)d_in[2];
    const int*   id_exp = (const int*)d_in[3];
    const int*   id_red = (const int*)d_in[4];
    const float* W_rbf  = (const float*)d_in[5];
    const float* W_sbf  = (const float*)d_in[6];
    const float* W_ji   = (const float*)d_in[7];
    const float* b_ji   = (const float*)d_in[8];
    const float* W_kj   = (const float*)d_in[9];
    const float* b_kj   = (const float*)d_in[10];
    const float* W_bil  = (const float*)d_in[11];
    const float* rbW    = (const float*)d_in[12];
    const float* rbB    = (const float*)d_in[13];
    const float* finW   = (const float*)d_in[14];
    const float* finB   = (const float*)d_in[15];
    const float* raW    = (const float*)d_in[16];
    const float* raB    = (const float*)d_in[17];
    float* out = (float*)d_out;

    const int NE = in_sizes[0] / FDIM;
    const int NT = in_sizes[3];

    char* ws = (char*)d_ws;
    size_t off = 0;
    float*  agg  = (float*)(ws + off);  off += (size_t)NE * FDIM * 4;   // x2 / segment sum
    float*  buf1 = (float*)(ws + off);  off += (size_t)NE * FDIM * 4;   // ping buffer
    __bf16* xkjg = (__bf16*)(ws + off); off += (size_t)NE * FDIM * 2;   // gathered A source
    float*  sbfp = (float*)(ws + off);  off += (size_t)NT * 8 * 4;      // sbf projection
    __bf16* wbf  = (__bf16*)(ws + off); off += (size_t)9 * 16384 * 2;   // 9 bf16 128x128 W
    __bf16* w3   = (__bf16*)(ws + off);                                 // 1024x128 bf16

    // --- weight conversion (bf16) ---
    const float* mats[9] = { W_ji, W_kj, rbW, rbW + 16384, finW,
                             raW, raW + 16384, raW + 32768, raW + 49152 };
    for (int i = 0; i < 9; ++i)
        bfcvt_kernel<<<64, 256, 0, stream>>>(mats[i], wbf + (size_t)i * 16384, 16384);
    w3prep_kernel<<<512, 256, 0, stream>>>(W_bil, w3);

    const dim3 dgrid(NE / 256);
    // x_kj = silu(x@W_kj+b)           -> buf1
    dense_silu_kernel<<<dgrid, 256, 0, stream>>>(x, wbf + 16384, b_kj, nullptr, buf1);
    // x_ji = silu(x@W_ji+b)           -> agg (init for segment sum)
    dense_silu_kernel<<<dgrid, 256, 0, stream>>>(x, wbf, b_ji, nullptr, agg);
    // x_kj * (rbf@W_rbf) -> bf16
    mul_rbf_bf16_kernel<<<(NE * 128) / 256, 256, 0, stream>>>(buf1, rbf, W_rbf, xkjg);
    // sbf_p
    sbf_proj_kernel<<<NT / 256, 256, 0, stream>>>(sbf, W_sbf, sbfp);
    // bilinear + scatter: agg += msg
    bilinear_scatter_kernel<<<1024, 256, 1024 * 72 * 2, stream>>>(
        xkjg, sbfp, id_exp, id_red, w3, agg, NT);
    // residual-before (NB=1)
    dense_silu_kernel<<<dgrid, 256, 0, stream>>>(agg,  wbf + 2 * 16384, rbB,        nullptr, buf1);
    dense_silu_kernel<<<dgrid, 256, 0, stream>>>(buf1, wbf + 3 * 16384, rbB + 128,  agg,     out);
    // final dense + skip: h = x + silu(x2@finW+fb)
    dense_silu_kernel<<<dgrid, 256, 0, stream>>>(out,  wbf + 4 * 16384, finB,       x,       buf1);
    // residual-after (NA=2)
    dense_silu_kernel<<<dgrid, 256, 0, stream>>>(buf1, wbf + 5 * 16384, raB,        nullptr, agg);
    dense_silu_kernel<<<dgrid, 256, 0, stream>>>(agg,  wbf + 6 * 16384, raB + 128,  buf1,    out);
    dense_silu_kernel<<<dgrid, 256, 0, stream>>>(out,  wbf + 7 * 16384, raB + 256,  nullptr, agg);
    dense_silu_kernel<<<dgrid, 256, 0, stream>>>(agg,  wbf + 8 * 16384, raB + 384,  out,     out);
}
